// Corr_40647570489454
// MI455X (gfx1250) — matrix operand order repaired and checked by reading, not compiled, not run on hardware
//
#include <hip/hip_runtime.h>
#include <stdint.h>

typedef _Float16 f16;
typedef __attribute__((ext_vector_type(16))) _Float16 v16h;
typedef __attribute__((ext_vector_type(8)))  float    v8f;
typedef __attribute__((ext_vector_type(4)))  int      v4i;

#if __has_builtin(__builtin_amdgcn_global_load_async_to_lds_b128) && __has_builtin(__builtin_amdgcn_s_wait_asynccnt)
#define USE_ASYNC 1
#else
#define USE_ASYNC 0
#endif

__device__ __forceinline__ void cp16(void* lds, const void* gl) {
#if USE_ASYNC
  __builtin_amdgcn_global_load_async_to_lds_b128(
      (__attribute__((address_space(1))) v4i*)gl,
      (__attribute__((address_space(3))) v4i*)lds, 0, 0);
#else
  *(uint4*)lds = *(const uint4*)gl;
#endif
}

__device__ __forceinline__ void stage_wait() {
#if USE_ASYNC
  __builtin_amdgcn_s_wait_asynccnt(0);
#endif
  __syncthreads();
}

// ---------------- K0: L2-normalize, emit f16 GEMM operands ----------------
// Apn[b][pix][c]  (row-major M x K, transposed from input)
// Bqn[b][c][pix]  (row-major K x N, same as input layout)
__global__ __launch_bounds__(256) void k_norm(const float* __restrict__ xp,
                                              const float* __restrict__ xq,
                                              f16* __restrict__ Apn,
                                              f16* __restrict__ Bqn) {
  int idx = blockIdx.x * 256 + threadIdx.x;       // 0..4095
  int which = idx >> 11;                          // 0: xp, 1: xq
  int b = (idx >> 10) & 1;
  int pix = idx & 1023;
  const float* src = (which ? xq : xp) + (size_t)b * 256 * 1024 + pix;
  float s = 0.f;
  for (int c = 0; c < 256; ++c) { float v = src[(size_t)c * 1024]; s += v * v; }
  float inv = 1.f / fmaxf(sqrtf(s), 1e-12f);
  if (which == 0) {
    f16* dst = Apn + (size_t)b * 1024 * 256 + (size_t)pix * 256;
    for (int c = 0; c < 256; ++c) dst[c] = (f16)(src[(size_t)c * 1024] * inv);
  } else {
    f16* dst = Bqn + (size_t)b * 256 * 1024 + pix;
    for (int c = 0; c < 256; ++c) dst[(size_t)c * 1024] = (f16)(src[(size_t)c * 1024] * inv);
  }
}

// ---------------- K1: pack conv1 weights to [K=96][N=16] f16 ----------------
__global__ __launch_bounds__(256) void k_w1pack(const float* __restrict__ W1,
                                                f16* __restrict__ w1p) {
  int idx = blockIdx.x * 256 + threadIdx.x;
  if (idx >= 96 * 16) return;
  int k = idx >> 4, co = idx & 15;
  float v = (k < 81) ? W1[co * 81 + k] : 0.f;
  w1p[k * 16 + co] = (f16)v;
}

// ---------------- K2: cost volume GEMM via WMMA ----------------
// aff[b][p=1024][q=1024] f16.  Block: 64(M) x 32(N), 8 waves = 4x2 of 16x16.
__global__ __launch_bounds__(256) void k_gemm(const f16* __restrict__ A,
                                              const f16* __restrict__ B,
                                              f16* __restrict__ aff) {
  __shared__ f16 As[64 * 32];
  __shared__ f16 Bs[32 * 32];
  int tid = threadIdx.x;
  int lane = tid & 31, wave = tid >> 5;
  int wm = wave >> 1, wn = wave & 1;
  int M0 = blockIdx.x * 64, N0 = blockIdx.y * 32;
  int b = blockIdx.z;
  const f16* Ab = A + (size_t)b * 1024 * 256;
  const f16* Bb = B + (size_t)b * 256 * 1024;
  int rowM = lane & 15, half = lane >> 4;
  v8f c = {};
  for (int kk = 0; kk < 256; kk += 32) {
    { int r = tid >> 2, cg = (tid & 3) * 8;
      cp16(&As[r * 32 + cg], &Ab[(size_t)(M0 + r) * 256 + kk + cg]); }
    if (tid < 128) {
      int r = tid >> 2, cg = (tid & 3) * 8;
      cp16(&Bs[r * 32 + cg], &Bb[(size_t)(kk + r) * 1024 + N0 + cg]);
    }
    stage_wait();
    v16h a, bf;
#pragma unroll
    for (int e = 0; e < 16; ++e) {
      int k = (e < 8 ? e : e + 8) + half * 8;       // ISA 16-bit A-frag K order
      a[e]  = As[(wm * 16 + rowM) * 32 + k];
      bf[e] = Bs[k * 32 + wn * 16 + rowM];          // B: lane = N col, e = K (same order as A)
    }
    c = __builtin_amdgcn_wmma_f32_16x16x32_f16(false, a, false, bf,
                                               (short)0, c, false, false);
    __syncthreads();
  }
  int N = lane & 15, mb = (lane >> 4) * 8;
#pragma unroll
  for (int r = 0; r < 8; ++r) {
    aff[(size_t)b * 1024 * 1024 + (size_t)(M0 + wm * 16 + mb + r) * 1024 +
        (N0 + wn * 16 + N)] = (f16)c[r];
  }
}

// ---------------- K3: conv1 (1->16 ch, 3^4 taps) as implicit-GEMM WMMA -----
// One block per (b,l,d). LDS holds 9 halo'd (l',d') planes of aff (34x34 f16)
// plus the packed 96x16 weight panel. Each wave: 16 pixels x 16 cout, K=96.
__global__ __launch_bounds__(256) void k_conv1(const f16* __restrict__ aff,
                                               const f16* __restrict__ w1p,
                                               const float* __restrict__ B1,
                                               f16* __restrict__ x1) {
  __shared__ f16 affs[9 * 34 * 34];
  __shared__ f16 wl[96 * 16];
  int tid = threadIdx.x;
  int blk = blockIdx.x;
  int b = blk >> 10, rem = blk & 1023;
  int l = rem >> 5, d = rem & 31;
  const f16* affb = aff + (size_t)b * 1024 * 1024;
  for (int idx = tid; idx < 9 * 34 * 34; idx += 256) {
    int pi = idx / (34 * 34), r2 = idx % (34 * 34);
    int hh = r2 / 34, ww = r2 % 34;
    int lp = l + pi / 3 - 1, dp = d + pi % 3 - 1;
    int hp = hh - 1, wp = ww - 1;
    f16 v = (f16)0.f;
    if (lp >= 0 && lp < 32 && dp >= 0 && dp < 32 &&
        hp >= 0 && hp < 32 && wp >= 0 && wp < 32)
      v = affb[(size_t)(lp * 32 + dp) * 1024 + hp * 32 + wp];
    affs[idx] = v;
  }
  for (int idx = tid; idx < 96 * 16; idx += 256) wl[idx] = w1p[idx];
  __syncthreads();

  int lane = tid & 31, wave = tid >> 5;
  int rowM = lane & 15, half = lane >> 4;
  int co = lane & 15;
  float bias = B1[16 + co];                         // l-dependent conv4d bias
  if (l > 0)  bias += B1[co];
  if (l < 31) bias += B1[32 + co];

  for (int it = 0; it < 8; ++it) {
    int T = wave * 8 + it;                          // 64 tiles of 16 pixels
    int h = T >> 1, w = ((T & 1) << 4) + rowM;
    v8f c = {};
#pragma unroll
    for (int ks = 0; ks < 3; ++ks) {
      int k0 = ks * 32;
      v16h a, bf;
#pragma unroll
      for (int e = 0; e < 16; ++e) {
        int t = k0 + (e < 8 ? e : e + 8) + half * 8;
        f16 av = (f16)0.f;
        if (t < 81) {
          int pi = t / 9;                           // i*3 + j
          int r9 = t % 9;
          int kk = r9 / 3, m = r9 % 3;
          av = affs[pi * 34 * 34 + (h + kk) * 34 + (w + m)];
        }
        a[e]  = av;
        bf[e] = wl[t * 16 + co];
      }
      c = __builtin_amdgcn_wmma_f32_16x16x32_f16(false, a, false, bf,
                                                 (short)0, c, false, false);
    }
    int mb = (lane >> 4) * 8;
    size_t base = ((((size_t)b * 32 + l) * 32 + d) * 16 + co) * 1024;
#pragma unroll
    for (int r = 0; r < 8; ++r) {
      int pix = T * 16 + mb + r;
      x1[base + pix] = (f16)(c[r] + bias);
    }
  }
}

// ---------------- K4: conv2 (16->1 ch) VALU with LDS slab ----------------
__global__ __launch_bounds__(256) void k_conv2(const f16* __restrict__ x1,
                                               const float* __restrict__ W2,
                                               const float* __restrict__ B2,
                                               float* __restrict__ xc) {
  __shared__ f16 slab[16 * 34 * 34];
  __shared__ float w2l[16 * 81];
  int tid = threadIdx.x;
  int blk = blockIdx.x;
  int b = blk >> 10, rem = blk & 1023;
  int l = rem >> 5, d = rem & 31;
  for (int idx = tid; idx < 16 * 81; idx += 256) w2l[idx] = W2[idx];
  float acc[4] = {0.f, 0.f, 0.f, 0.f};
  int w = tid & 31, h0 = tid >> 5;
  for (int pi = 0; pi < 9; ++pi) {
    int lp = l + pi / 3 - 1, dp = d + pi % 3 - 1;
    __syncthreads();
    bool valid = (lp >= 0 && lp < 32 && dp >= 0 && dp < 32);
    const f16* src = x1 + ((((size_t)b * 32 + (valid ? lp : 0)) * 32 +
                            (valid ? dp : 0)) * 16) * 1024;
    for (int idx = tid; idx < 16 * 34 * 34; idx += 256) {
      int co = idx / (34 * 34), r2 = idx % (34 * 34);
      int hh = r2 / 34, ww = r2 % 34;
      int hp = hh - 1, wp = ww - 1;
      f16 v = (f16)0.f;
      if (valid && hp >= 0 && hp < 32 && wp >= 0 && wp < 32)
        v = src[(size_t)co * 1024 + hp * 32 + wp];
      slab[idx] = v;
    }
    __syncthreads();
    int i = pi / 3, j = pi % 3;
#pragma unroll
    for (int q = 0; q < 4; ++q) {
      int h = h0 + 8 * q;
      float a = 0.f;
      for (int co = 0; co < 16; ++co) {
        const float* wrow = &w2l[co * 81 + i * 27 + j * 9];
        const f16* srow = &slab[co * 34 * 34 + h * 34 + w];
#pragma unroll
        for (int kk = 0; kk < 3; ++kk)
#pragma unroll
          for (int m = 0; m < 3; ++m)
            a += (float)srow[kk * 34 + m] * wrow[kk * 3 + m];
      }
      acc[q] += a;
    }
  }
  float bias = B2[1] + (l > 0 ? B2[0] : 0.f) + (l < 31 ? B2[2] : 0.f);
  size_t base = ((size_t)b * 1024 + (size_t)(l * 32 + d)) * 1024;
  for (int q = 0; q < 4; ++q) {
    int h = h0 + 8 * q;
    xc[base + h * 32 + w] = acc[q] + bias;
  }
}

// ---------------- K5/K6: top-3 (sorted desc) over rows / columns ----------
__global__ __launch_bounds__(256) void k_topk_rows(const float* __restrict__ xc,
                                                   float* __restrict__ out) {
  int idx = blockIdx.x * 256 + threadIdx.x;        // valp: per (b,p) scan q
  if (idx >= 2048) return;
  int b = idx >> 10, p = idx & 1023;
  const float* row = xc + ((size_t)b * 1024 + p) * 1024;
  float t0 = -3.4e38f, t1 = t0, t2 = t0;
  for (int q = 0; q < 1024; ++q) {
    float v = row[q];
    if (v > t0) { t2 = t1; t1 = t0; t0 = v; }
    else if (v > t1) { t2 = t1; t1 = v; }
    else if (v > t2) { t2 = v; }
  }
  out[(size_t)(b * 3 + 0) * 1024 + p] = t0;
  out[(size_t)(b * 3 + 1) * 1024 + p] = t1;
  out[(size_t)(b * 3 + 2) * 1024 + p] = t2;
}

__global__ __launch_bounds__(256) void k_topk_cols(const float* __restrict__ xc,
                                                   float* __restrict__ out) {
  int idx = blockIdx.x * 256 + threadIdx.x;        // valq: per (b,q) scan p
  if (idx >= 2048) return;
  int b = idx >> 10, q = idx & 1023;
  const float* col = xc + (size_t)b * 1024 * 1024 + q;
  float t0 = -3.4e38f, t1 = t0, t2 = t0;
  for (int p = 0; p < 1024; ++p) {
    float v = col[(size_t)p * 1024];
    if (v > t0) { t2 = t1; t1 = t0; t0 = v; }
    else if (v > t1) { t2 = t1; t1 = v; }
    else if (v > t2) { t2 = v; }
  }
  out[(size_t)(b * 3 + 0) * 1024 + q] = t0;
  out[(size_t)(b * 3 + 1) * 1024 + q] = t1;
  out[(size_t)(b * 3 + 2) * 1024 + q] = t2;
}

extern "C" void kernel_launch(void* const* d_in, const int* in_sizes, int n_in,
                              void* d_out, int out_size, void* d_ws, size_t ws_size,
                              hipStream_t stream) {
  (void)in_sizes; (void)n_in; (void)out_size; (void)ws_size;
  const float* xp = (const float*)d_in[0];
  const float* xq = (const float*)d_in[1];
  const float* W1 = (const float*)d_in[2];
  const float* B1 = (const float*)d_in[3];
  const float* W2 = (const float*)d_in[4];
  const float* B2 = (const float*)d_in[5];
  float* out = (float*)d_out;
  char* ws = (char*)d_ws;
  // workspace layout (bytes): Apn 1MB | Bqn 1MB | aff 4MB | w1p | x1 64MB | xc 8MB
  f16*   Apn = (f16*)(ws + 0);
  f16*   Bqn = (f16*)(ws + (1u << 20));
  f16*   aff = (f16*)(ws + (2u << 20));
  f16*   w1p = (f16*)(ws + (6u << 20));
  f16*   x1  = (f16*)(ws + (7u << 20));
  float* xc  = (float*)(ws + (7u << 20) + (64u << 20));

  k_norm<<<dim3(16), dim3(256), 0, stream>>>(xp, xq, Apn, Bqn);
  k_w1pack<<<dim3(6), dim3(256), 0, stream>>>(W1, w1p);
  k_gemm<<<dim3(16, 32, 2), dim3(256), 0, stream>>>(Apn, Bqn, aff);
  k_conv1<<<dim3(2048), dim3(256), 0, stream>>>(aff, w1p, B1, x1);
  k_conv2<<<dim3(2048), dim3(256), 0, stream>>>(x1, W2, B2, xc);
  k_topk_rows<<<dim3(8), dim3(256), 0, stream>>>(xc, out);
  k_topk_cols<<<dim3(8), dim3(256), 0, stream>>>(xc, out + 6144);
}
